// TransformerNet_24189255811791
// MI455X (gfx1250) — compile-verified, matrix operand order checked by reading, not run on hardware
//
#include <hip/hip_runtime.h>
#include <hip/hip_bf16.h>
#include <stddef.h>

// ---------------------------------------------------------------- constants
#define NNODES 10000
#define NEDGES 80000
#define NGRAPH 64
#define NHEADS 8
#define OUTDIM 10
#define FMAX   512        // widest activation width
#define MPAD   10112      // NNODES rounded up to 128 (79 * 128)

// ---------------------------------------------------------------- types
typedef __attribute__((ext_vector_type(16))) __bf16         bf16x16;
typedef __attribute__((ext_vector_type(8)))  float          f32x8;
typedef __attribute__((ext_vector_type(8)))  unsigned short us8;
typedef __attribute__((ext_vector_type(16))) unsigned short us16;
typedef __attribute__((ext_vector_type(4)))  int            v4i;

union BFPack { us16 u; bf16x16 v; };

#define AS1 __attribute__((address_space(1)))
#define AS3 __attribute__((address_space(3)))

// ---------------------------------------------------------------- async LDS path
#if defined(__HIP_DEVICE_COMPILE__) && __has_builtin(__builtin_amdgcn_global_load_async_to_lds_b128)
#define HAVE_ASYNC 1
#else
#define HAVE_ASYNC 0
#endif

#if HAVE_ASYNC
#define ASYNC_COPY16(gp, lp)                                              \
  __builtin_amdgcn_global_load_async_to_lds_b128(                         \
      (AS1 v4i*)(gp), (AS3 v4i*)(lp), 0, 0)
# if __has_builtin(__builtin_amdgcn_s_wait_asynccnt)
#  define ASYNC_WAIT(n) __builtin_amdgcn_s_wait_asynccnt(n)
# else
#  define ASYNC_WAIT(n) asm volatile("s_wait_asynccnt %0" ::"i"(n))
# endif
#else
#define ASYNC_COPY16(gp, lp) (*(uint4*)(lp) = *(const uint4*)(gp))
#define ASYNC_WAIT(n) ((void)0)
#endif

static __device__ __forceinline__ unsigned short f2bf(float f) {
  union { float f; unsigned u; } c; c.f = f;
  unsigned r = c.u + 0x7fffu + ((c.u >> 16) & 1u);   // round-to-nearest-even
  return (unsigned short)(r >> 16);
}

static __device__ __forceinline__ float atomicMaxF(float* addr, float val) {
  unsigned* ua = (unsigned*)addr;
  unsigned old = *ua;
  while (true) {
    float f = __uint_as_float(old);
    if (f >= val) break;
    unsigned assumed = old;
    old = atomicCAS(ua, assumed, __float_as_uint(val));
    if (old == assumed) break;
  }
  return __uint_as_float(old);
}

// ---------------------------------------------------------------- utility kernels
__global__ void k_fill(float* __restrict__ p, float v, size_t n) {
  size_t i = (size_t)blockIdx.x * blockDim.x + threadIdx.x;
  if (i < n) p[i] = v;
}

// bf16 convert with zero padding: out[i] = i < n_valid ? bf(in[i]) : 0
__global__ void k_f32_to_bf16(const float* __restrict__ in,
                              unsigned short* __restrict__ out,
                              size_t n_valid, size_t n_total) {
  size_t i = (size_t)blockIdx.x * blockDim.x + threadIdx.x;
  if (i < n_total) out[i] = (i < n_valid) ? f2bf(in[i]) : (unsigned short)0;
}

// Wt[n*K + k] = bf16( W[k*Nfull + col0 + n] )   (transpose + slice + convert)
__global__ void k_w_transpose_bf16(const float* __restrict__ W,
                                   unsigned short* __restrict__ Wt,
                                   int K, int Nfull, int col0, int ncols) {
  int t = blockIdx.x * blockDim.x + threadIdx.x;
  if (t >= K * ncols) return;
  int n = t % ncols, k = t / ncols;
  Wt[(size_t)n * K + k] = f2bf(W[(size_t)k * Nfull + col0 + n]);
}

// ---------------------------------------------------------------- WMMA GEMM (LDS-staged)
// C[MPAD x Nn] = A[MPAD x K](bf16 row-major) * Bt[Nn x K](bf16) + bias
// Block: 256 thr = 8 waves, tile 128(M) x 64(N); wave = 32x32 (2x2 fragments).
// K stepped by 32 with double-buffered async global->LDS staging.
#define LROW 40   // LDS row pitch in ushorts (32 data + 8 pad, 80B)
__global__ __launch_bounds__(256)
void k_gemm_wmma_bf16(const unsigned short* __restrict__ A,
                      const unsigned short* __restrict__ Bt,
                      const float* __restrict__ bias,
                      float* __restrict__ C,
                      int K, int Nn) {
  __shared__ unsigned short As[2][128 * LROW];
  __shared__ unsigned short Bs[2][64 * LROW];

  const int tm0 = blockIdx.y * 128;
  const int tn0 = blockIdx.x * 64;
  const int tid = threadIdx.x;
  const int lane = tid & 31;
  const int w    = tid >> 5;
  const int wm   = w & 3;        // wave M sub-tile (0..3) -> rows wm*32
  const int wn   = w >> 2;       // wave N sub-tile (0..1) -> cols wn*32
  const int lidx = lane & 15;
  const int half = lane >> 4;    // K-octet select (ISA 16-bit A/B layout)

  // stage one K-slice (k0..k0+31) of A-block and B-block into LDS buffer `buf`
  auto stage = [&](int buf, int k0) {
    #pragma unroll
    for (int t = 0; t < 3; ++t) {
      int cid = tid + t * 256;               // 0..767, exactly 3 chunks/thread
      if (cid < 512) {                       // A: 128 rows x 4 chunks
        int row = cid >> 2, j = cid & 3;
        const unsigned short* g = A + (size_t)(tm0 + row) * K + k0 + j * 8;
        unsigned short* l = &As[buf][row * LROW + j * 8];
        ASYNC_COPY16(g, l);
      } else {                               // B: 64 rows x 4 chunks
        int c2 = cid - 512;
        int row = c2 >> 2, j = c2 & 3;
        const unsigned short* g = Bt + (size_t)(tn0 + row) * K + k0 + j * 8;
        unsigned short* l = &Bs[buf][row * LROW + j * 8];
        ASYNC_COPY16(g, l);
      }
    }
  };

  f32x8 acc00 = {}, acc01 = {}, acc10 = {}, acc11 = {};
  const int nk = K >> 5;

  stage(0, 0);
  int cur = 0;
  for (int kt = 0; kt < nk; ++kt) {
    if (kt + 1 < nk) { stage(cur ^ 1, (kt + 1) << 5); ASYNC_WAIT(3); }
    else             { ASYNC_WAIT(0); }
    __syncthreads();

    // load fragments from LDS (two 16B loads each)
    BFPack a0, a1, b0, b1;
    {
      const unsigned short* p0 = &As[cur][(wm * 32 + lidx) * LROW + half * 8];
      const unsigned short* p1 = &As[cur][(wm * 32 + 16 + lidx) * LROW + half * 8];
      const unsigned short* q0 = &Bs[cur][(wn * 32 + lidx) * LROW + half * 8];
      const unsigned short* q1 = &Bs[cur][(wn * 32 + 16 + lidx) * LROW + half * 8];
      us8 lo, hi;
      lo = *(const us8*)p0; hi = *(const us8*)(p0 + 16);
      #pragma unroll
      for (int i = 0; i < 8; ++i) { a0.u[i] = lo[i]; a0.u[i + 8] = hi[i]; }
      lo = *(const us8*)p1; hi = *(const us8*)(p1 + 16);
      #pragma unroll
      for (int i = 0; i < 8; ++i) { a1.u[i] = lo[i]; a1.u[i + 8] = hi[i]; }
      lo = *(const us8*)q0; hi = *(const us8*)(q0 + 16);
      #pragma unroll
      for (int i = 0; i < 8; ++i) { b0.u[i] = lo[i]; b0.u[i + 8] = hi[i]; }
      lo = *(const us8*)q1; hi = *(const us8*)(q1 + 16);
      #pragma unroll
      for (int i = 0; i < 8; ++i) { b1.u[i] = lo[i]; b1.u[i + 8] = hi[i]; }
    }
    acc00 = __builtin_amdgcn_wmma_f32_16x16x32_bf16(false, a0.v, false, b0.v, (short)0, acc00, false, false);
    acc01 = __builtin_amdgcn_wmma_f32_16x16x32_bf16(false, a0.v, false, b1.v, (short)0, acc01, false, false);
    acc10 = __builtin_amdgcn_wmma_f32_16x16x32_bf16(false, a1.v, false, b0.v, (short)0, acc10, false, false);
    acc11 = __builtin_amdgcn_wmma_f32_16x16x32_bf16(false, a1.v, false, b1.v, (short)0, acc11, false, false);

    __syncthreads();   // all waves done reading buf before it is restaged
    cur ^= 1;
  }

  // store 2x2 fragments (+bias); MPAD padding => no row guards needed
  const int col0 = tn0 + wn * 32 + lidx;
  const int col1 = col0 + 16;
  const float bv0 = bias ? bias[col0] : 0.0f;
  const float bv1 = bias ? bias[col1] : 0.0f;
  const int r0 = tm0 + wm * 32 + half * 8;
  #pragma unroll
  for (int v = 0; v < 8; ++v) {
    C[(size_t)(r0 + v) * Nn + col0]      = acc00[v] + bv0;
    C[(size_t)(r0 + v) * Nn + col1]      = acc01[v] + bv1;
    C[(size_t)(r0 + 16 + v) * Nn + col0] = acc10[v] + bv0;
    C[(size_t)(r0 + 16 + v) * Nn + col1] = acc11[v] + bv1;
  }
}

// ---------------------------------------------------------------- edge attention
__global__ void k_edge_logits(const float* __restrict__ q, const float* __restrict__ k,
                              const int* __restrict__ src, const int* __restrict__ dst,
                              float* __restrict__ a, int E, int c, float scale) {
  int e    = blockIdx.x * (blockDim.x >> 5) + (threadIdx.x >> 5);
  int lane = threadIdx.x & 31;
  if (e >= E) return;
  int s = src[e], d = dst[e];
  const float* qp = q + (size_t)d * c;
  const float* kp = k + (size_t)s * c;
  float acc = 0.f;
  for (int i = lane; i < c; i += 32) acc += qp[i] * kp[i];
  #pragma unroll
  for (int off = 16; off; off >>= 1) acc += __shfl_xor(acc, off, 32);
  if (lane == 0) a[e] = acc * scale;
}

__global__ void k_edge_max(const float* __restrict__ a, const int* __restrict__ dst,
                           float* __restrict__ m, int E) {
  int e = blockIdx.x * blockDim.x + threadIdx.x;
  if (e < E) atomicMaxF(&m[dst[e]], a[e]);
}

__global__ void k_edge_expsum(float* __restrict__ a, const int* __restrict__ dst,
                              const float* __restrict__ m, float* __restrict__ s, int E) {
  int e = blockIdx.x * blockDim.x + threadIdx.x;
  if (e < E) {
    int d = dst[e];
    float ev = __expf(a[e] - m[d]);
    a[e] = ev;
    atomicAdd(&s[d], ev);
  }
}

__global__ void k_edge_scatter(const float* __restrict__ a, const float* __restrict__ s,
                               const float* __restrict__ v,
                               const int* __restrict__ src, const int* __restrict__ dst,
                               float* __restrict__ accum, int c) {
  int e = blockIdx.x;
  int sn = src[e], dn = dst[e];
  float alpha = a[e] / (s[dn] + 1e-16f);
  const float* vp = v + (size_t)sn * c;
  float* op = accum + (size_t)dn * c;
  for (int i = threadIdx.x; i < c; i += blockDim.x)
    atomicAdd(&op[i], alpha * vp[i]);
}

__global__ void k_finalize_elu(const float* __restrict__ accum, float* __restrict__ hio,
                               size_t n) {
  size_t i = (size_t)blockIdx.x * blockDim.x + threadIdx.x;
  if (i < n) {
    float x = hio[i] + accum[i] * (1.0f / (float)NHEADS);
    hio[i] = x > 0.f ? x : (__expf(x) - 1.0f);
  }
}

// ---------------------------------------------------------------- pooling / head
__global__ void k_gate(const float* __restrict__ h, const float* __restrict__ wg,
                       const float* __restrict__ bg, float* __restrict__ gate,
                       int N, int c) {
  int n    = blockIdx.x * (blockDim.x >> 5) + (threadIdx.x >> 5);
  int lane = threadIdx.x & 31;
  if (n >= N) return;
  const float* hp = h + (size_t)n * c;
  float acc = 0.f;
  for (int i = lane; i < c; i += 32) acc += hp[i] * wg[i];
  #pragma unroll
  for (int off = 16; off; off >>= 1) acc += __shfl_xor(acc, off, 32);
  if (lane == 0) gate[n] = acc + bg[0];
}

__global__ void k_gate_max(const float* __restrict__ gate, const int* __restrict__ batch,
                           float* __restrict__ gm, int N) {
  int n = blockIdx.x * blockDim.x + threadIdx.x;
  if (n < N) atomicMaxF(&gm[batch[n]], gate[n]);
}

__global__ void k_gate_expsum(float* __restrict__ gate, const int* __restrict__ batch,
                              const float* __restrict__ gm, float* __restrict__ gs, int N) {
  int n = blockIdx.x * blockDim.x + threadIdx.x;
  if (n < N) {
    int b = batch[n];
    float ev = __expf(gate[n] - gm[b]);
    gate[n] = ev;
    atomicAdd(&gs[b], ev);
  }
}

__global__ void k_pool_scatter(const float* __restrict__ h, const float* __restrict__ gate,
                               const float* __restrict__ gs, const int* __restrict__ batch,
                               float* __restrict__ g, int c) {
  int n = blockIdx.x;
  int b = batch[n];
  float alpha = gate[n] / (gs[b] + 1e-16f);
  const float* hp = h + (size_t)n * c;
  float* gp = g + (size_t)b * c;
  for (int i = threadIdx.x; i < c; i += blockDim.x)
    atomicAdd(&gp[i], alpha * hp[i]);
}

__global__ void k_final_fc(const float* __restrict__ g, const float* __restrict__ wfc,
                           const float* __restrict__ bfc, float* __restrict__ out,
                           int G, int c, int O) {
  int t = blockIdx.x * blockDim.x + threadIdx.x;
  if (t >= G * O) return;
  int gi = t / O, o = t % O;
  float acc = bfc[o];
  const float* gp = g + (size_t)gi * c;
  for (int k2 = 0; k2 < c; ++k2) acc += gp[k2] * wfc[(size_t)k2 * O + o];
  out[t] = acc;
}

// ---------------------------------------------------------------- launcher
extern "C" void kernel_launch(void* const* d_in, const int* in_sizes, int n_in,
                              void* d_out, int out_size, void* d_ws, size_t ws_size,
                              hipStream_t stream) {
  const float* x     = (const float*)d_in[0];
  const int*   ei    = (const int*)d_in[1];
  const int*   batch = (const int*)d_in[2];
  const int*   src   = ei;
  const int*   dst   = ei + NEDGES;
  const float* wg    = (const float*)d_in[27];
  const float* bg    = (const float*)d_in[28];
  const float* wfc   = (const float*)d_in[29];
  const float* bfc   = (const float*)d_in[30];

  struct LayerP { const float *wq,*bq,*wk,*bk,*wv,*bv,*ws,*bs; int fin, c; };
  LayerP L[3];
  for (int l = 0; l < 3; ++l) {
    int base = 3 + l * 8;
    L[l].wq = (const float*)d_in[base+0]; L[l].bq = (const float*)d_in[base+1];
    L[l].wk = (const float*)d_in[base+2]; L[l].bk = (const float*)d_in[base+3];
    L[l].wv = (const float*)d_in[base+4]; L[l].bv = (const float*)d_in[base+5];
    L[l].ws = (const float*)d_in[base+6]; L[l].bs = (const float*)d_in[base+7];
  }
  L[0].fin = 128; L[0].c = 512;
  L[1].fin = 512; L[1].c = 256;
  L[2].fin = 256; L[2].c = 128;

  // ---------------- workspace carve (activation buffers padded to MPAD rows)
  char* wsp = (char*)d_ws;
  size_t used = 0;
  auto alloc = [&](size_t bytes) -> void* {
    void* r = wsp + used;
    used += (bytes + 255) & ~(size_t)255;
    return r;
  };
  float* bufA           = (float*)alloc((size_t)MPAD * FMAX * 4);
  float* bufB           = (float*)alloc((size_t)MPAD * FMAX * 4);
  unsigned short* hbf   = (unsigned short*)alloc((size_t)MPAD * FMAX * 2);
  unsigned short* wtbf  = (unsigned short*)alloc((size_t)512 * 512 * 2);
  float* qb             = (float*)alloc((size_t)MPAD * FMAX * 4);
  float* kb             = (float*)alloc((size_t)MPAD * FMAX * 4);
  float* vb             = (float*)alloc((size_t)MPAD * FMAX * 4);
  float* accum          = (float*)alloc((size_t)MPAD * FMAX * 4);
  float* alog           = (float*)alloc((size_t)NEDGES * 4);
  float* mbuf           = (float*)alloc((size_t)NNODES * 4);
  float* sbuf           = (float*)alloc((size_t)NNODES * 4);
  float* gate           = (float*)alloc((size_t)NNODES * 4);
  float* gm             = (float*)alloc((size_t)NGRAPH * 4);
  float* gs             = (float*)alloc((size_t)NGRAPH * 4);
  float* gpool          = (float*)alloc((size_t)NGRAPH * 128 * 4);
  if (used > ws_size) return;   // workspace too small: bail deterministically

  auto gemm = [&](const unsigned short* A, const unsigned short* Bt, const float* bias,
                  float* C, int K, int Nn) {
    dim3 grid(Nn / 64, MPAD / 128);
    k_gemm_wmma_bf16<<<grid, 256, 0, stream>>>(A, Bt, bias, C, K, Nn);
  };
  auto fill = [&](float* p, float v, size_t n) {
    k_fill<<<(unsigned)((n + 255) / 256), 256, 0, stream>>>(p, v, n);
  };

  const float* hin = x;
  float* outbufs[3] = { bufA, bufB, bufA };

  for (int l = 0; l < 3; ++l) {
    const int fin = L[l].fin, c = L[l].c;
    float* hnext = outbufs[l];

    // activations -> bf16 (zero-padded to MPAD rows)
    {
      size_t nv = (size_t)NNODES * fin, nt = (size_t)MPAD * fin;
      k_f32_to_bf16<<<(unsigned)((nt + 255) / 256), 256, 0, stream>>>(hin, hbf, nv, nt);
    }
    // skip path: hnext = hin @ ws + bs
    {
      int tot = fin * c;
      k_w_transpose_bf16<<<(tot + 255) / 256, 256, 0, stream>>>(L[l].ws, wtbf, fin, c, 0, c);
      gemm(hbf, wtbf, L[l].bs, hnext, fin, c);
    }
    // attention heads
    fill(accum, 0.f, (size_t)NNODES * c);
    const float scale = rsqrtf((float)c);
    for (int h = 0; h < NHEADS; ++h) {
      int tot = fin * c;
      k_w_transpose_bf16<<<(tot + 255) / 256, 256, 0, stream>>>(L[l].wq, wtbf, fin, NHEADS * c, h * c, c);
      gemm(hbf, wtbf, L[l].bq + h * c, qb, fin, c);
      k_w_transpose_bf16<<<(tot + 255) / 256, 256, 0, stream>>>(L[l].wk, wtbf, fin, NHEADS * c, h * c, c);
      gemm(hbf, wtbf, L[l].bk + h * c, kb, fin, c);
      k_w_transpose_bf16<<<(tot + 255) / 256, 256, 0, stream>>>(L[l].wv, wtbf, fin, NHEADS * c, h * c, c);
      gemm(hbf, wtbf, L[l].bv + h * c, vb, fin, c);

      k_edge_logits<<<(NEDGES + 7) / 8, 256, 0, stream>>>(qb, kb, src, dst, alog, NEDGES, c, scale);
      fill(mbuf, -1e30f, NNODES);
      fill(sbuf, 0.f, NNODES);
      k_edge_max<<<(NEDGES + 255) / 256, 256, 0, stream>>>(alog, dst, mbuf, NEDGES);
      k_edge_expsum<<<(NEDGES + 255) / 256, 256, 0, stream>>>(alog, dst, mbuf, sbuf, NEDGES);
      k_edge_scatter<<<NEDGES, 256, 0, stream>>>(alog, sbuf, vb, src, dst, accum, c);
    }
    // h = elu(skip + mean_heads)
    {
      size_t n = (size_t)NNODES * c;
      k_finalize_elu<<<(unsigned)((n + 255) / 256), 256, 0, stream>>>(accum, hnext, n);
    }
    hin = hnext;
  }

  // ---------------- global attention pooling + FC (hin = [N,128])
  const int c3 = 128;
  k_gate<<<(NNODES + 7) / 8, 256, 0, stream>>>(hin, wg, bg, gate, NNODES, c3);
  fill(gm, -1e30f, NGRAPH);
  fill(gs, 0.f, NGRAPH);
  k_gate_max<<<(NNODES + 255) / 256, 256, 0, stream>>>(gate, batch, gm, NNODES);
  k_gate_expsum<<<(NNODES + 255) / 256, 256, 0, stream>>>(gate, batch, gm, gs, NNODES);
  fill(gpool, 0.f, (size_t)NGRAPH * c3);
  k_pool_scatter<<<NNODES, 128, 0, stream>>>(hin, gate, gs, batch, gpool, c3);
  k_final_fc<<<(NGRAPH * OUTDIM + 255) / 256, 256, 0, stream>>>(
      gpool, wfc, bfc, (float*)d_out, NGRAPH, c3, OUTDIM);
}
